// SeqSelfAttention_21010980012181
// MI455X (gfx1250) — matrix operand order repaired and checked by reading, not compiled, not run on hardware
//
#include <hip/hip_runtime.h>
#include <hip/hip_bf16.h>

// ---------------------------------------------------------------------------
// SeqSelfAttention (additive / Bahdanau, local band width 64, centered)
// B=2, L=1024, D=256, U=64.  Band sparsity: exp(-10000+..) == 0.0f in fp32,
// so only j in [i-32, i+32) matters.  GEMM stages use v_wmma_f32_16x16x32_f16;
// the band-logit stage uses native v_tanh_f32 (TRANS, co-executes with WMMA).
// ---------------------------------------------------------------------------

#define BB 2
#define LL 1024
#define DD 256
#define UU 64
#define NTILES ((BB * LL) / 16)   // 128

typedef _Float16 h16;
typedef h16   v8h  __attribute__((ext_vector_type(8)));
typedef h16   v16h __attribute__((ext_vector_type(16)));
typedef float v8f  __attribute__((ext_vector_type(8)));

__device__ __forceinline__ v16h cat16(v8h a, v8h b) {
  return __builtin_shufflevector(a, b, 0,1,2,3,4,5,6,7,8,9,10,11,12,13,14,15);
}

// Native CDNA5 V_TANH_F32 when available (gfx1250 TRANS op); libcall fallback.
__device__ __forceinline__ float fast_tanh(float v) {
#if defined(__AMDGCN__) && __has_builtin(__builtin_amdgcn_tanhf)
  return __builtin_amdgcn_tanhf(v);
#else
  return tanhf(v);
#endif
}

// A-operand (16x32 f16, MxK): lane<16 -> row=lane,    K in {k0..k0+7,  k0+16..k0+23}
//                             lane>=16 -> row=lane-16, K in {k0+8..15,  k0+24..31}
__device__ __forceinline__ v16h load_A(const h16* rowbase, int k0, int lane) {
  int off = (lane < 16) ? 0 : 8;
  v8h lo = *(const v8h*)(rowbase + k0 + off);
  v8h hi = *(const v8h*)(rowbase + k0 + 16 + off);
  return cat16(lo, hi);
}

// B-operand (32x16 f16, KxN): lane n (=lane&15) holds column n; lanes 0-15
// cover K = k0..k0+7, k0+16..k0+23; lanes 16-31 cover K = k0+8..15, k0+24..31.
__device__ __forceinline__ v16h load_B(const h16* colbase /*row-major over K*/, int k0, int lane) {
  int off = (lane < 16) ? 0 : 8;
  v8h lo = *(const v8h*)(colbase + k0 + off);
  v8h hi = *(const v8h*)(colbase + k0 + off + 16);
  return cat16(lo, hi);
}

__device__ __forceinline__ v8f wmma_f16(v16h a, v16h b, v8f c) {
  return __builtin_amdgcn_wmma_f32_16x16x32_f16(false, a, false, b, (short)0, c, false, false);
}

// ---------------------------------------------------------------------------
// Kernel 1: q = x*Wt ; k' = x*Wx + bh      (stored f32 into scratch)
// grid = 128 tiles of 16 rows, block = 256 (8 waves)
// ---------------------------------------------------------------------------
#define XA_STR 264   // halfs per row (256 + 8 pad: conflict-free, 16B aligned)
#define WT_STR 264

__global__ __launch_bounds__(256)
void proj_kernel(const float* __restrict__ x, const float* __restrict__ Wt,
                 const float* __restrict__ Wx, const float* __restrict__ bh,
                 float* __restrict__ qbuf, float* __restrict__ kbuf) {
  __shared__ alignas(16) h16 xA[16 * XA_STR];
  __shared__ alignas(16) h16 wT[2][UU * WT_STR];
  __shared__ float bhs[UU];

  const int g0  = blockIdx.x * 16;     // global row (b*L + i)
  const int tid = threadIdx.x;

  for (int e = tid; e < 16 * DD; e += 256) {
    int r = e >> 8, d = e & (DD - 1);
    xA[r * XA_STR + d] = (h16)x[(size_t)(g0 + r) * DD + d];
  }
  for (int e = tid; e < DD * UU; e += 256) {
    int d = e / UU, u = e % UU;
    wT[0][u * WT_STR + d] = (h16)Wt[e];
    wT[1][u * WT_STR + d] = (h16)Wx[e];
  }
  if (tid < UU) bhs[tid] = bh[tid];
  __syncthreads();

  const int wave = tid >> 5, lane = tid & 31;
  const int which = wave >> 2;            // 0 -> q (Wt), 1 -> k' (Wx)
  const int n0 = (wave & 3) * 16;
  const h16* wbase = &wT[which][0];
  const int n = lane & 15;

  v8f acc = {0.f,0.f,0.f,0.f,0.f,0.f,0.f,0.f};
  for (int ks = 0; ks < 8; ++ks) {
    int k0 = ks * 32;
    v16h a = load_A(&xA[n * XA_STR], k0, lane);           // row = lane&15
    v16h b = load_B(&wbase[(n0 + n) * WT_STR], k0, lane); // col = lane&15
    acc = wmma_f16(a, b, acc);
  }

  float* obase = (which == 0) ? qbuf : kbuf;
  const float bias = (which == 1) ? bhs[n0 + n] : 0.0f;   // uniform, branchless
  const int mhi = (lane < 16) ? 0 : 8;
  for (int r = 0; r < 8; ++r)
    obase[(size_t)(g0 + r + mhi) * UU + n0 + n] = acc[r] + bias;
}

// ---------------------------------------------------------------------------
// Kernel 2: banded additive attention + v = a*x (WMMA)
// grid = 128 tiles (b, i0), block = 256 (8 waves)
// ---------------------------------------------------------------------------
#define JT 96        // padded j-window: j in [i0-32, i0+64), band needs < 80
#define XS_STR 104   // halfs (96 + 8): 208B rows -> 16B aligned, conflict-free
#define AS_STR 104
#define KB_STR 66    // floats (64 + 2): kills 64-stride bank conflicts

__global__ __launch_bounds__(256)
void attn_kernel(const float* __restrict__ x, const float* __restrict__ qbuf,
                 const float* __restrict__ kbuf, const float* __restrict__ Wa,
                 float* __restrict__ out) {
  __shared__ alignas(16) h16 xsT[DD * XS_STR];    // x^T tile, f16   (53.2 KB)
  __shared__ alignas(16) h16 as_[16 * AS_STR];    // attn weights f16 (3.3 KB)
  __shared__ float qs[16 * UU];                   // q tile           (4 KB)
  __shared__ float kbs[JT * KB_STR];              // k'+bh tile      (25.3 KB)
  __shared__ float es[16 * JT];                   // logits           (6 KB)
  __shared__ float was[UU];

  const int t  = blockIdx.x;
  const int b  = t >> 6;              // t / (L/16)
  const int i0 = (t & 63) * 16;
  const int tid = threadIdx.x;
  const int gbase = b * LL;

  // ---- stage x^T (f16, zero-filled outside [0,L)) ----
  for (int e = tid; e < JT * DD; e += 256) {
    int jj = e >> 8, d = e & (DD - 1);           // d == tid each iter: coalesced
    int j = i0 - 32 + jj;
    float v = (j >= 0 && j < LL) ? x[(size_t)(gbase + j) * DD + d] : 0.0f;
    xsT[d * XS_STR + jj] = (h16)v;
  }
  // ---- stage q, k' (f32) ----
  for (int e = tid; e < 16 * UU; e += 256) {
    qs[e] = qbuf[(size_t)(gbase + i0) * UU + e];
  }
  for (int e = tid; e < JT * UU; e += 256) {
    int jj = e / UU, u = e % UU;
    int j = i0 - 32 + jj;
    kbs[jj * KB_STR + u] = (j >= 0 && j < LL) ? kbuf[(size_t)(gbase + j) * UU + u] : 0.0f;
  }
  if (tid < UU) was[tid] = Wa[tid];
  __syncthreads();

  // ---- banded logits: e[ii][jj] = sum_u tanh(q+k')*Wa  (band only) ----
  for (int e = tid; e < 16 * JT; e += 256) {
    int ii = e / JT, jj = e % JT;
    int j = i0 - 32 + jj;
    bool band = (jj >= ii) && (jj < ii + 64) && (j >= 0) && (j < LL);
    float s = -1.0e30f;
    if (band) {
      s = 0.0f;
      const float* qrow = &qs[ii * UU];
      const float* krow = &kbs[jj * KB_STR];
      #pragma unroll 8
      for (int u = 0; u < UU; ++u) s += fast_tanh(qrow[u] + krow[u]) * was[u];
    }
    es[ii * JT + jj] = s;   // +ba omitted: uniform shift cancels in softmax
  }
  __syncthreads();

  // ---- row softmax (16 rows, exact-zero outside band via exp underflow) ----
  if (tid < 16) {
    float* row = &es[tid * JT];
    float m = -1.0e30f;
    for (int jj = 0; jj < JT; ++jj) m = fmaxf(m, row[jj]);
    float sum = 0.0f;
    for (int jj = 0; jj < JT; ++jj) { float p = __expf(row[jj] - m); row[jj] = p; sum += p; }
    float inv = 1.0f / sum;
    for (int jj = 0; jj < JT; ++jj) as_[tid * AS_STR + jj] = (h16)(row[jj] * inv);
  }
  __syncthreads();

  // ---- v = a @ x : [16 x 96] * [96 x 16]-tiles via WMMA, 2 d-tiles/wave ----
  const int wave = tid >> 5, lane = tid & 31;
  const int n = lane & 15;
  const int mhi = (lane < 16) ? 0 : 8;
  for (int pass = 0; pass < 2; ++pass) {
    int d0 = (wave + 8 * pass) * 16;
    v8f acc = {0.f,0.f,0.f,0.f,0.f,0.f,0.f,0.f};
    for (int ks = 0; ks < 3; ++ks) {
      int k0 = ks * 32;
      v16h a = load_A(&as_[n * AS_STR], k0, lane);
      v16h bm = load_B(&xsT[(d0 + n) * XS_STR], k0, lane);
      acc = wmma_f16(a, bm, acc);
    }
    for (int r = 0; r < 8; ++r)
      out[(size_t)(gbase + i0 + r + mhi) * DD + d0 + n] = acc[r];
  }
}

// ---------------------------------------------------------------------------
extern "C" void kernel_launch(void* const* d_in, const int* in_sizes, int n_in,
                              void* d_out, int out_size, void* d_ws, size_t ws_size,
                              hipStream_t stream) {
  const float* x  = (const float*)d_in[0];
  const float* Wt = (const float*)d_in[1];
  const float* Wx = (const float*)d_in[2];
  const float* bh = (const float*)d_in[3];
  const float* Wa = (const float*)d_in[4];
  // ba (d_in[5]) is a uniform logit shift -> cancels in softmax; unused.
  float* out = (float*)d_out;

  float* qbuf = (float*)d_ws;                         // [B*L, U] f32
  float* kbuf = qbuf + (size_t)BB * LL * UU;          // [B*L, U] f32 (bh folded)

  proj_kernel<<<NTILES, 256, 0, stream>>>(x, Wt, Wx, bh, qbuf, kbuf);
  attn_kernel<<<NTILES, 256, 0, stream>>>(x, qbuf, kbuf, Wa, out);
}
